// GroupedQueryAttention_12120397709900
// MI455X (gfx1250) — matrix operand order repaired and checked by reading, not compiled, not run on hardware
//
#include <hip/hip_runtime.h>
#include <hip/hip_bf16.h>
#include <math.h>
#include <stdint.h>

typedef __bf16 bf16;
typedef __bf16   v16bf __attribute__((ext_vector_type(16)));
typedef float    v8f   __attribute__((ext_vector_type(8)));
typedef unsigned u32x4 __attribute__((ext_vector_type(4)));

union FragU { v16bf v; u32x4 u[2]; };

#define WMMA_BF16(a, b, c) \
  __builtin_amdgcn_wmma_f32_16x16x32_bf16(false, (a), false, (b), (short)0, (c), false, false)

#define BATCH 2
#define SEQ   2048
#define DIMSZ 4096
#define NHEAD 32
#define NKVH  8
#define HDIM  128
#define NQKV  6144          // NH*HD + 2*NKV*HD
#define SCALE 0.08838834764831843f   // 1/sqrt(128)

// ---------------------------------------------------------------------------
// CDNA5 async global->LDS copy path (ASYNCcnt-tracked), with safe fallback.
#ifndef __has_builtin
#define __has_builtin(x) 0
#endif

#if __has_builtin(__builtin_amdgcn_global_load_async_to_lds_b128)
#define HAVE_ASYNC_LDS 1
#else
#define HAVE_ASYNC_LDS 0
#endif

typedef int v4i_t __attribute__((ext_vector_type(4)));
typedef __attribute__((address_space(1))) v4i_t gv4i_t;   // global (AS1)
typedef __attribute__((address_space(3))) v4i_t lv4i_t;   // LDS (AS3)

__device__ __forceinline__ void async_copy_b128(const void* g, void* l) {
#if HAVE_ASYNC_LDS
  __builtin_amdgcn_global_load_async_to_lds_b128(
      (gv4i_t*)(uintptr_t)g, (lv4i_t*)l, 0, 0);
#else
  *(u32x4*)l = *(const u32x4*)g;
#endif
}

__device__ __forceinline__ void wait_async() {
#if __has_builtin(__builtin_amdgcn_s_wait_asynccnt)
  __builtin_amdgcn_s_wait_asynccnt(0);
#elif HAVE_ASYNC_LDS
  asm volatile("s_wait_asynccnt 0x0" ::: "memory");
#endif
}

// ---------------------------------------------------------------------------
// elementwise fp32 -> bf16
__global__ void cvt_bf16_kernel(const float* __restrict__ src, bf16* __restrict__ dst, int n) {
  int i = blockIdx.x * blockDim.x + threadIdx.x;
  if (i < n) dst[i] = (bf16)src[i];
}

// transpose-convert: src is K x N (row-major), dst is N x K (row-major) bf16
__global__ void tcvt_bf16_kernel(const float* __restrict__ src, bf16* __restrict__ dst,
                                 int K, int N) {
  int i = blockIdx.x * blockDim.x + threadIdx.x;
  if (i >= K * N) return;
  int k = i % K;
  int n = i / K;
  dst[(size_t)n * K + k] = (bf16)src[(size_t)k * N + n];
}

// ---------------------------------------------------------------------------
// Generic bf16 GEMM: C(MxN fp32) = A(MxK bf16, row-major) * Bt(NxK bf16, row-major)^T
// Block tile 128x128, 256 threads = 8 waves (2 in M x 4 in N), each wave 64x32.
// Double-buffered LDS stages filled by async global->LDS copies.
__global__ __launch_bounds__(256) void gemm_bf16_kernel(
    const bf16* __restrict__ A, const bf16* __restrict__ Bt, float* __restrict__ C,
    int M, int N, int K) {
  __shared__ bf16 sA[2][128 * 40];   // 32 k + 8 pad (16B-aligned rows)
  __shared__ bf16 sB[2][128 * 40];

  const int tid  = threadIdx.x;
  const int lane = tid & 31;
  const int w    = tid >> 5;
  const int half = lane >> 4;
  const int l16  = lane & 15;
  const int wm   = w & 1;        // 0..1
  const int wn   = w >> 1;       // 0..3
  const int m0   = blockIdx.y * 128;
  const int n0   = blockIdx.x * 128;

  v8f acc[4][2];
#pragma unroll
  for (int i = 0; i < 4; ++i)
#pragma unroll
    for (int j = 0; j < 2; ++j)
#pragma unroll
      for (int r = 0; r < 8; ++r) acc[i][j][r] = 0.0f;

  auto copy_stage = [&](int buf, int k0) {
#pragma unroll
    for (int t = 0; t < 2; ++t) {        // 512 x b128 per tile pair, 2+2 per thread
      int q   = tid + t * 256;
      int row = q >> 2;
      int seg = q & 3;
      async_copy_b128(&A[(size_t)(m0 + row) * K + k0 + seg * 8],
                      &sA[buf][row * 40 + seg * 8]);
      async_copy_b128(&Bt[(size_t)(n0 + row) * K + k0 + seg * 8],
                      &sB[buf][row * 40 + seg * 8]);
    }
  };

  // prologue: fill stage 0
  copy_stage(0, 0);
  wait_async();
  __syncthreads();

  int buf = 0;
  for (int k0 = 0; k0 < K; k0 += 32) {
    int nk = k0 + 32;
    if (nk < K) {
      copy_stage(buf ^ 1, nk);
      if (nk + 32 < K) {      // prefetch 2 stages ahead (global_prefetch_b8)
        int pr = tid & 127;
        const bf16* pp = (tid < 128)
            ? &A[(size_t)(m0 + pr) * K + nk + 32]
            : &Bt[(size_t)(n0 + pr) * K + nk + 32];
        __builtin_prefetch(pp, 0, 0);
      }
    }

    FragU af[4], bfg[2];
#pragma unroll
    for (int i = 0; i < 4; ++i) {
      int r = wm * 64 + i * 16 + l16;
      af[i].u[0] = *(const u32x4*)&sA[buf][r * 40 + half * 8];
      af[i].u[1] = *(const u32x4*)&sA[buf][r * 40 + 16 + half * 8];
    }
#pragma unroll
    for (int j = 0; j < 2; ++j) {
      int c = wn * 32 + j * 16 + l16;
      bfg[j].u[0] = *(const u32x4*)&sB[buf][c * 40 + half * 8];
      bfg[j].u[1] = *(const u32x4*)&sB[buf][c * 40 + 16 + half * 8];
    }
#pragma unroll
    for (int i = 0; i < 4; ++i)
#pragma unroll
      for (int j = 0; j < 2; ++j)
        acc[i][j] = WMMA_BF16(af[i].v, bfg[j].v, acc[i][j]);

    wait_async();       // our async writes into buf^1 are done
    __syncthreads();    // everyone done reading buf / writing buf^1
    buf ^= 1;
  }

#pragma unroll
  for (int i = 0; i < 4; ++i)
#pragma unroll
    for (int j = 0; j < 2; ++j) {
      int n = n0 + wn * 32 + j * 16 + l16;
#pragma unroll
      for (int r = 0; r < 8; ++r) {
        int m = m0 + wm * 64 + i * 16 + r + 8 * half;
        C[(size_t)m * N + n] = acc[i][j][r];
      }
    }
}

// ---------------------------------------------------------------------------
// RoPE + scatter: reads fused QKV fp32, writes q/k bf16 (b,h,s,d), new_k/new_v
// fp32 to d_out in (b,s,h,d), and V transposed bf16 (b,h,d,s) for PV matmul.
__global__ void rope_scatter_kernel(const float* __restrict__ Cqkv,
                                    const float* __restrict__ fc,
                                    float* __restrict__ outK, float* __restrict__ outV,
                                    bf16* __restrict__ qb, bf16* __restrict__ kb,
                                    bf16* __restrict__ vt) {
  const int NQp = BATCH * SEQ * NHEAD * 64;
  const int NKp = BATCH * SEQ * NKVH * 64;
  int idx = blockIdx.x * blockDim.x + threadIdx.x;
  if (idx < NQp) {
    int i = idx & 63;
    int h = (idx >> 6) % NHEAD;
    int s = ((idx >> 6) / NHEAD) % SEQ;
    int b = (idx >> 6) / NHEAD / SEQ;
    float cs = fc[(s * 64 + i) * 2 + 0];
    float sn = fc[(s * 64 + i) * 2 + 1];
    size_t row = (size_t)(b * SEQ + s) * NQKV;
    float xr = Cqkv[row + h * HDIM + 2 * i];
    float xi = Cqkv[row + h * HDIM + 2 * i + 1];
    float orr = xr * cs - xi * sn;
    float oii = xr * sn + xi * cs;
    size_t qo = ((size_t)(b * NHEAD + h) * SEQ + s) * HDIM + 2 * i;
    qb[qo]     = (bf16)orr;
    qb[qo + 1] = (bf16)oii;
  } else if (idx < NQp + NKp) {
    int j = idx - NQp;
    int i = j & 63;
    int h = (j >> 6) % NKVH;
    int s = ((j >> 6) / NKVH) % SEQ;
    int b = (j >> 6) / NKVH / SEQ;
    float cs = fc[(s * 64 + i) * 2 + 0];
    float sn = fc[(s * 64 + i) * 2 + 1];
    size_t row = (size_t)(b * SEQ + s) * NQKV;
    float xr = Cqkv[row + NHEAD * HDIM + h * HDIM + 2 * i];
    float xi = Cqkv[row + NHEAD * HDIM + h * HDIM + 2 * i + 1];
    float orr = xr * cs - xi * sn;
    float oii = xr * sn + xi * cs;
    size_t ko = ((size_t)(b * SEQ + s) * NKVH + h) * HDIM + 2 * i;
    outK[ko]     = orr;
    outK[ko + 1] = oii;
    size_t kbo = ((size_t)(b * NKVH + h) * SEQ + s) * HDIM + 2 * i;
    kb[kbo]     = (bf16)orr;
    kb[kbo + 1] = (bf16)oii;
  } else if (idx < NQp + 2 * NKp) {
    int j = idx - NQp - NKp;
    int i = j & 63;
    int h = (j >> 6) % NKVH;
    int s = ((j >> 6) / NKVH) % SEQ;
    int b = (j >> 6) / NKVH / SEQ;
    size_t row = (size_t)(b * SEQ + s) * NQKV;
    float xr = Cqkv[row + (NHEAD + NKVH) * HDIM + h * HDIM + 2 * i];
    float xi = Cqkv[row + (NHEAD + NKVH) * HDIM + h * HDIM + 2 * i + 1];
    size_t vo = ((size_t)(b * SEQ + s) * NKVH + h) * HDIM + 2 * i;
    outV[vo]     = xr;
    outV[vo + 1] = xi;
    size_t vt0 = ((size_t)(b * NKVH + h) * HDIM + 2 * i) * SEQ + s;
    vt[vt0]       = (bf16)xr;
    vt[vt0 + SEQ] = (bf16)xi;
  }
}

// ---------------------------------------------------------------------------
// Flash attention: block = one (b,h) x 128 q rows; 8 waves x 16 q rows.
// K tile (64x128) and Vt tile (128x64) staged in LDS via async copies.
__global__ __launch_bounds__(256) void attn_kernel(
    const bf16* __restrict__ Q, const bf16* __restrict__ Kc,
    const bf16* __restrict__ Vt, bf16* __restrict__ ctx) {
  __shared__ bf16 lK[64 * 136];      // 64 keys x 128 hd (+8 pad)
  __shared__ bf16 lV[128 * 72];      // 128 hd x 64 keys (+8 pad)
  __shared__ bf16 lP[8 * 16 * 72];   // per-wave 16 q x 64 keys (+8 pad)

  const int tid  = threadIdx.x;
  const int lane = tid & 31;
  const int w    = tid >> 5;
  const int half = lane >> 4;
  const int l16  = lane & 15;
  const int b    = blockIdx.z;
  const int h    = blockIdx.y;
  const int qt   = blockIdx.x;
  const int hkv  = h >> 2;           // N_REP = 4
  const int qrow0 = qt * 128 + w * 16;

  // Q fragments held in registers for the whole K loop
  const bf16* qbase = Q + ((size_t)(b * NHEAD + h) * SEQ + qrow0 + l16) * HDIM;
  FragU aq[4];
#pragma unroll
  for (int c = 0; c < 4; ++c) {
    aq[c].u[0] = *(const u32x4*)(qbase + c * 32 + half * 8);
    aq[c].u[1] = *(const u32x4*)(qbase + c * 32 + 16 + half * 8);
  }

  v8f o[8];
  float mrow[8], lrow[8];
#pragma unroll
  for (int t = 0; t < 8; ++t) {
#pragma unroll
    for (int r = 0; r < 8; ++r) o[t][r] = 0.0f;
    mrow[t] = -1e30f;
    lrow[t] = 0.0f;
  }

  const bf16* kg = Kc + ((size_t)(b * NKVH + hkv) * SEQ) * HDIM;
  const bf16* vg = Vt + ((size_t)(b * NKVH + hkv) * HDIM) * SEQ;
  bf16* pmine = lP + w * 16 * 72;

  for (int kbk = 0; kbk < SEQ / 64; ++kbk) {
#pragma unroll
    for (int t = 0; t < 4; ++t) {            // K tile: 1024 x b128 async
      int q = tid + t * 256;
      int row = q >> 3, seg = q & 7;
      async_copy_b128(&kg[(size_t)(kbk * 64 + row) * HDIM + seg * 8],
                      &lK[row * 136 + seg * 8]);
    }
#pragma unroll
    for (int t = 0; t < 4; ++t) {            // Vt tile: 1024 x b128 async
      int q = tid + t * 256;
      int row = q >> 3, seg = q & 7;
      async_copy_b128(&vg[(size_t)row * SEQ + kbk * 64 + seg * 8],
                      &lV[row * 72 + seg * 8]);
    }
    wait_async();
    __syncthreads();

    // scores: 16 q x 64 keys per wave
    v8f sacc[4];
#pragma unroll
    for (int j = 0; j < 4; ++j) {
#pragma unroll
      for (int r = 0; r < 8; ++r) sacc[j][r] = 0.0f;
      int kr = j * 16 + l16;
#pragma unroll
      for (int c = 0; c < 4; ++c) {
        FragU bk;
        bk.u[0] = *(const u32x4*)&lK[kr * 136 + c * 32 + half * 8];
        bk.u[1] = *(const u32x4*)&lK[kr * 136 + c * 32 + 16 + half * 8];
        sacc[j] = WMMA_BF16(aq[c].v, bk.v, sacc[j]);
      }
    }

    // online softmax (row = r + 8*half, matching C layout)
    float p[4][8], bm[8];
#pragma unroll
    for (int r = 0; r < 8; ++r) bm[r] = -1e30f;
#pragma unroll
    for (int j = 0; j < 4; ++j)
#pragma unroll
      for (int r = 0; r < 8; ++r) {
        float sv = sacc[j][r] * SCALE;
        p[j][r] = sv;
        bm[r] = fmaxf(bm[r], sv);
      }
#pragma unroll
    for (int m = 1; m < 16; m <<= 1)
#pragma unroll
      for (int r = 0; r < 8; ++r) bm[r] = fmaxf(bm[r], __shfl_xor(bm[r], m, 32));

    float alpha[8], rs[8];
#pragma unroll
    for (int r = 0; r < 8; ++r) {
      float mn = fmaxf(mrow[r], bm[r]);
      alpha[r] = __expf(mrow[r] - mn);
      mrow[r] = mn;
      rs[r] = 0.0f;
    }
#pragma unroll
    for (int j = 0; j < 4; ++j)
#pragma unroll
      for (int r = 0; r < 8; ++r) {
        float e = __expf(p[j][r] - mrow[r]);
        p[j][r] = e;
        rs[r] += e;
      }
#pragma unroll
    for (int m = 1; m < 16; m <<= 1)
#pragma unroll
      for (int r = 0; r < 8; ++r) rs[r] += __shfl_xor(rs[r], m, 32);
#pragma unroll
    for (int r = 0; r < 8; ++r) lrow[r] = lrow[r] * alpha[r] + rs[r];
#pragma unroll
    for (int t = 0; t < 8; ++t)
#pragma unroll
      for (int r = 0; r < 8; ++r) o[t][r] *= alpha[r];

    // P: C layout -> LDS (row major) -> A fragments (per-wave, DS in-order)
#pragma unroll
    for (int j = 0; j < 4; ++j)
#pragma unroll
      for (int r = 0; r < 8; ++r)
        pmine[(r + 8 * half) * 72 + j * 16 + l16] = (bf16)p[j][r];

#pragma unroll
    for (int c = 0; c < 2; ++c) {
      FragU ap;
      ap.u[0] = *(const u32x4*)&pmine[l16 * 72 + c * 32 + half * 8];
      ap.u[1] = *(const u32x4*)&pmine[l16 * 72 + c * 32 + 16 + half * 8];
#pragma unroll
      for (int t = 0; t < 8; ++t) {
        FragU bv;
        int vr = t * 16 + l16;
        bv.u[0] = *(const u32x4*)&lV[vr * 72 + c * 32 + half * 8];
        bv.u[1] = *(const u32x4*)&lV[vr * 72 + c * 32 + 16 + half * 8];
        o[t] = WMMA_BF16(ap.v, bv.v, o[t]);
      }
    }
    __syncthreads();
  }

  float inv[8];
#pragma unroll
  for (int r = 0; r < 8; ++r) inv[r] = 1.0f / lrow[r];
  size_t crow = (size_t)(b * SEQ + qrow0);
#pragma unroll
  for (int t = 0; t < 8; ++t)
#pragma unroll
    for (int r = 0; r < 8; ++r)
      ctx[(crow + r + 8 * half) * (NHEAD * HDIM) + h * HDIM + t * 16 + l16] =
          (bf16)(o[t][r] * inv[r]);
}

// ---------------------------------------------------------------------------
extern "C" void kernel_launch(void* const* d_in, const int* in_sizes, int n_in,
                              void* d_out, int out_size, void* d_ws, size_t ws_size,
                              hipStream_t stream) {
  const float* x  = (const float*)d_in[0];
  const float* fc = (const float*)d_in[1];
  const float* wq = (const float*)d_in[2];
  const float* wk = (const float*)d_in[3];
  const float* wv = (const float*)d_in[4];
  const float* wo = (const float*)d_in[5];

  float* out0 = (float*)d_out;                        // (B,S,DIM)
  float* outK = out0 + (size_t)BATCH * SEQ * DIMSZ;   // (B,S,NKV,HD)
  float* outV = outK + (size_t)BATCH * SEQ * NKVH * HDIM;

  char* ws = (char*)d_ws;
  bf16*  xb    = (bf16*)ws;                           // 33,554,432 B
  bf16*  wqkvT = (bf16*)(ws + 33554432ull);           // 50,331,648 B (6144 x 4096)
  bf16*  woT   = (bf16*)(ws + 83886080ull);           // 33,554,432 B
  float* Cqkv  = (float*)(ws + 117440512ull);         // 100,663,296 B (4096 x 6144)
  bf16*  qb    = (bf16*)(ws + 218103808ull);          // 33,554,432 B
  bf16*  kb    = (bf16*)(ws + 251658240ull);          // 8,388,608 B
  bf16*  vt    = (bf16*)(ws + 260046848ull);          // 8,388,608 B
  bf16*  ctx   = (bf16*)Cqkv;                         // reuse (Cqkv dead after rope)

  // 1) precision conversion + weight transposes
  cvt_bf16_kernel<<<65536, 256, 0, stream>>>(x, xb, BATCH * SEQ * DIMSZ);
  tcvt_bf16_kernel<<<65536, 256, 0, stream>>>(wq, wqkvT, DIMSZ, NHEAD * HDIM);
  tcvt_bf16_kernel<<<16384, 256, 0, stream>>>(wk, wqkvT + (size_t)4096 * 4096, DIMSZ, NKVH * HDIM);
  tcvt_bf16_kernel<<<16384, 256, 0, stream>>>(wv, wqkvT + (size_t)5120 * 4096, DIMSZ, NKVH * HDIM);
  tcvt_bf16_kernel<<<65536, 256, 0, stream>>>(wo, woT, NHEAD * HDIM, DIMSZ);

  // 2) fused QKV projection: (4096 x 4096) @ (4096 x 6144)
  gemm_bf16_kernel<<<dim3(48, 32), 256, 0, stream>>>(
      xb, wqkvT, Cqkv, BATCH * SEQ, NQKV, DIMSZ);

  // 3) RoPE + layout scatter (also emits new_k / new_v fp32)
  rope_scatter_kernel<<<49152, 256, 0, stream>>>(Cqkv, fc, outK, outV, qb, kb, vt);

  // 4) flash attention -> bf16 ctx (b,s, h*HD)
  attn_kernel<<<dim3(16, 32, 2), 256, 0, stream>>>(qb, kb, vt, ctx);

  // 5) output projection: (4096 x 4096) @ (4096 x 4096) -> fp32 out
  gemm_bf16_kernel<<<dim3(32, 32), 256, 0, stream>>>(
      ctx, woT, out0, BATCH * SEQ, DIMSZ, DIMSZ);
}
